// GCN_30803505447557
// MI455X (gfx1250) — matrix operand order repaired and checked by reading, not compiled, not run on hardware
//
#include <hip/hip_runtime.h>
#include <cstdint>

#define NND   10000     // nodes
#define NE    320000    // edges
#define DF    256       // feature dim (in & hidden)
#define NG    64        // graphs
#define DOUT  16        // classifier out

typedef _Float16 v16h __attribute__((ext_vector_type(16)));
typedef _Float16 v8h  __attribute__((ext_vector_type(8)));
typedef float    v8f  __attribute__((ext_vector_type(8)));

// ---------------- utility kernels ----------------

__global__ void zero_f32(float* __restrict__ p, int n) {
  int t = blockIdx.x * blockDim.x + threadIdx.x;
  if (t < n) p[t] = 0.0f;
}

__global__ void deg_kernel(const int* __restrict__ dst, float* __restrict__ cnt) {
  int e = blockIdx.x * blockDim.x + threadIdx.x;
  if (e < NE) unsafeAtomicAdd(&cnt[dst[e]], 1.0f);
}

__global__ void f2h_kernel(const float* __restrict__ s, _Float16* __restrict__ d, int n) {
  int t = blockIdx.x * blockDim.x + threadIdx.x;
  if (t < n) d[t] = (_Float16)s[t];
}

// Scatter-add x[src] into sum[dst]: ONE WAVE32 PER EDGE.
// Edge indices are loaded once per wave (uniform address -> single L2 request);
// each lane moves 8 contiguous floats (2x b128 gather) and issues 8
// global_atomic_add_f32. Feature matrix (10.2MB) is L2-resident (192MB L2).
__global__ __launch_bounds__(256)
void scatter_kernel(const float* __restrict__ feat,
                    const int* __restrict__ src,
                    const int* __restrict__ dst,
                    float* __restrict__ sum) {
  int wave = (blockIdx.x * blockDim.x + threadIdx.x) >> 5;
  int lane = threadIdx.x & 31;
  if (wave >= NE) return;
  int s = src[wave];
  int d = dst[wave];
  const float4* f = (const float4*)(feat + (size_t)s * DF) + lane * 2;
  float* o = sum + (size_t)d * DF + lane * 8;
  float4 v0 = f[0];
  float4 v1 = f[1];
  unsafeAtomicAdd(o + 0, v0.x);
  unsafeAtomicAdd(o + 1, v0.y);
  unsafeAtomicAdd(o + 2, v0.z);
  unsafeAtomicAdd(o + 3, v0.w);
  unsafeAtomicAdd(o + 4, v1.x);
  unsafeAtomicAdd(o + 5, v1.y);
  unsafeAtomicAdd(o + 6, v1.z);
  unsafeAtomicAdd(o + 7, v1.w);
}

// Mean-normalize the aggregation and emit both GEMM A-operands in f16.
__global__ void norm_cvt_kernel(const float* __restrict__ sum,
                                const float* __restrict__ cnt,
                                const float* __restrict__ self,
                                _Float16* __restrict__ aggrh,
                                _Float16* __restrict__ selfh) {
  int t = blockIdx.x * blockDim.x + threadIdx.x;
  if (t >= NND * DF) return;
  int i = t >> 8;            // node id (DF == 256)
  float c = cnt[i];
  c = (c < 1.0f) ? 1.0f : c;
  aggrh[t] = (_Float16)(sum[t] / c);
  selfh[t] = (_Float16)self[t];
}

// ---------------- fused SAGE GEMM (WMMA) ----------------
// out = relu( Aagg @ Wl^T + Aself @ Wr^T + bias ), all [10000 x 256]
// One wave32 per 16x16 output tile; two v_wmma_f32_16x16x32_f16 per K-step.
__global__ __launch_bounds__(256)
void sage_gemm_kernel(const _Float16* __restrict__ Aagg,
                      const _Float16* __restrict__ Aself,
                      const _Float16* __restrict__ Wl,   // [DF][DF] row-major (out x in)
                      const _Float16* __restrict__ Wr,
                      const float* __restrict__ bias,    // [DF]
                      float* __restrict__ out) {
  int wave = (blockIdx.x * blockDim.x + threadIdx.x) >> 5;
  int lane = threadIdx.x & 31;
  int mt = wave >> 4;        // 625 M-tiles
  int nt = wave & 15;        // 16  N-tiles
  int l15 = lane & 15;
  int hi  = lane >> 4;

  int arow = mt * 16 + l15;                    // A row held by this lane
  int col  = nt * 16 + l15;                    // B column held by this lane
  const _Float16* a0p = Aagg  + (size_t)arow * DF;
  const _Float16* a1p = Aself + (size_t)arow * DF;
  const _Float16* b0p = Wl + (size_t)col * DF; // B[k][col] = W[col][k]
  const _Float16* b1p = Wr + (size_t)col * DF;

  v8f acc = {};
  #pragma unroll
  for (int k0 = 0; k0 < DF; k0 += 32) {
    // A 16x32 f16 layout (ISA 7.12.2): lane-half 0 -> K {0..7,16..23}, half 1 -> K {8..15,24..31}
    union { v16h v; v8h h[2]; } a0, a1, b0, b1;
    a0.h[0] = *(const v8h*)(a0p + k0 + hi * 8);
    a0.h[1] = *(const v8h*)(a0p + k0 + 16 + hi * 8);
    a1.h[0] = *(const v8h*)(a1p + k0 + hi * 8);
    a1.h[1] = *(const v8h*)(a1p + k0 + 16 + hi * 8);
    // B 32x16 f16: same K order as A: lane-half 0 -> K {0..7,16..23}, half 1 -> K {8..15,24..31}
    b0.h[0] = *(const v8h*)(b0p + k0 + hi * 8);
    b0.h[1] = *(const v8h*)(b0p + k0 + 16 + hi * 8);
    b1.h[0] = *(const v8h*)(b1p + k0 + hi * 8);
    b1.h[1] = *(const v8h*)(b1p + k0 + 16 + hi * 8);
    acc = __builtin_amdgcn_wmma_f32_16x16x32_f16(false, a0.v, false, b0.v,
                                                 (short)0, acc, false, false);
    acc = __builtin_amdgcn_wmma_f32_16x16x32_f16(false, a1.v, false, b1.v,
                                                 (short)0, acc, false, false);
  }

  // C/D layout: VGPR r -> M = (hi?8:0)+r, N = l15
  float bn = bias[col];
  int rbase = mt * 16 + hi * 8;
  #pragma unroll
  for (int r = 0; r < 8; ++r) {
    float v = acc[r] + bn;
    v = (v > 0.0f) ? v : 0.0f;
    out[(size_t)(rbase + r) * DF + col] = v;
  }
}

// ---------------- pooling + classifier ----------------

// One wave32 per node: batch[i] loaded once per wave, each lane adds 8 floats.
__global__ __launch_bounds__(256)
void pool_kernel(const float* __restrict__ h,
                 const int* __restrict__ batch,
                 float* __restrict__ pooled) {
  int wave = (blockIdx.x * blockDim.x + threadIdx.x) >> 5;
  int lane = threadIdx.x & 31;
  if (wave >= NND) return;
  int g = batch[wave];
  const float4* f = (const float4*)(h + (size_t)wave * DF) + lane * 2;
  float* o = pooled + (size_t)g * DF + lane * 8;
  float4 v0 = f[0];
  float4 v1 = f[1];
  unsafeAtomicAdd(o + 0, v0.x);
  unsafeAtomicAdd(o + 1, v0.y);
  unsafeAtomicAdd(o + 2, v0.z);
  unsafeAtomicAdd(o + 3, v0.w);
  unsafeAtomicAdd(o + 4, v1.x);
  unsafeAtomicAdd(o + 5, v1.y);
  unsafeAtomicAdd(o + 6, v1.z);
  unsafeAtomicAdd(o + 7, v1.w);
}

__global__ void cls_kernel(const float* __restrict__ pooled,
                           const float* __restrict__ w,    // [DOUT][DF]
                           const float* __restrict__ b,    // [DOUT]
                           float* __restrict__ out) {      // [NG][DOUT]
  int t = blockIdx.x * blockDim.x + threadIdx.x;
  if (t >= NG * DOUT) return;
  int g = t >> 4;
  int o = t & 15;
  const float4* pr = (const float4*)(pooled + (size_t)g * DF);
  const float4* wr = (const float4*)(w + (size_t)o * DF);
  float s = 0.0f;
  #pragma unroll 4
  for (int k = 0; k < DF / 4; ++k) {
    float4 a = pr[k], c = wr[k];
    s += a.x * c.x + a.y * c.y + a.z * c.z + a.w * c.w;
  }
  s += b[o];
  out[t] = 1.0f / (1.0f + __expf(-s));
}

// ---------------- host-side launch ----------------

extern "C" void kernel_launch(void* const* d_in, const int* in_sizes, int n_in,
                              void* d_out, int out_size, void* d_ws, size_t ws_size,
                              hipStream_t stream) {
  const float* x    = (const float*)d_in[0];
  const int*   ei   = (const int*)d_in[1];   // [2][NE]: row0=src, row1=dst (int32)
  const int*   bat  = (const int*)d_in[2];
  const float* w1l  = (const float*)d_in[3];
  const float* b1l  = (const float*)d_in[4];
  const float* w1r  = (const float*)d_in[5];
  const float* w2l  = (const float*)d_in[6];
  const float* b2l  = (const float*)d_in[7];
  const float* w2r  = (const float*)d_in[8];
  const float* wcls = (const float*)d_in[9];
  const float* bcls = (const float*)d_in[10];
  float* out = (float*)d_out;

  const int* esrc = ei;
  const int* edst = ei + NE;

  // workspace carve (256B aligned)
  char* p = (char*)d_ws;
  auto carve = [&](size_t bytes) -> void* {
    void* r = (void*)p;
    p += (bytes + 255) & ~(size_t)255;
    return r;
  };
  float*    cnt    = (float*)carve((size_t)NND * 4);
  float*    sum    = (float*)carve((size_t)NND * DF * 4);   // also reused as h2
  float*    h1     = (float*)carve((size_t)NND * DF * 4);
  _Float16* aggrh  = (_Float16*)carve((size_t)NND * DF * 2);
  _Float16* selfh  = (_Float16*)carve((size_t)NND * DF * 2);
  _Float16* w1lh   = (_Float16*)carve((size_t)DF * DF * 2);
  _Float16* w1rh   = (_Float16*)carve((size_t)DF * DF * 2);
  _Float16* w2lh   = (_Float16*)carve((size_t)DF * DF * 2);
  _Float16* w2rh   = (_Float16*)carve((size_t)DF * DF * 2);
  float*    pooled = (float*)carve((size_t)NG * DF * 4);

  const int TB = 256;
  auto cdiv = [](long long a, long long b) { return (int)((a + b - 1) / b); };
  const int nfeat = NND * DF;                  // 2,560,000
  const int scatterBlocks = cdiv((long long)NE * 32, TB);   // wave per edge
  const int poolBlocks    = cdiv((long long)NND * 32, TB);  // wave per node

  // degrees (shared by both layers)
  zero_f32<<<cdiv(NND, TB), TB, 0, stream>>>(cnt, NND);
  deg_kernel<<<cdiv(NE, TB), TB, 0, stream>>>(edst, cnt);

  // weights -> f16 (once per call; deterministic)
  f2h_kernel<<<cdiv(DF * DF, TB), TB, 0, stream>>>(w1l, w1lh, DF * DF);
  f2h_kernel<<<cdiv(DF * DF, TB), TB, 0, stream>>>(w1r, w1rh, DF * DF);
  f2h_kernel<<<cdiv(DF * DF, TB), TB, 0, stream>>>(w2l, w2lh, DF * DF);
  f2h_kernel<<<cdiv(DF * DF, TB), TB, 0, stream>>>(w2r, w2rh, DF * DF);

  const int gemmBlocks = (NND / 16) * 16 * 32 / TB;  // 625*16 waves -> 1250 blocks

  // ---- layer 1 ----
  zero_f32<<<cdiv(nfeat, TB), TB, 0, stream>>>(sum, nfeat);
  scatter_kernel<<<scatterBlocks, TB, 0, stream>>>(x, esrc, edst, sum);
  norm_cvt_kernel<<<cdiv(nfeat, TB), TB, 0, stream>>>(sum, cnt, x, aggrh, selfh);
  sage_gemm_kernel<<<gemmBlocks, TB, 0, stream>>>(aggrh, selfh, w1lh, w1rh, b1l, h1);

  // ---- layer 2 ----
  zero_f32<<<cdiv(nfeat, TB), TB, 0, stream>>>(sum, nfeat);
  scatter_kernel<<<scatterBlocks, TB, 0, stream>>>(h1, esrc, edst, sum);
  norm_cvt_kernel<<<cdiv(nfeat, TB), TB, 0, stream>>>(sum, cnt, h1, aggrh, selfh);
  sage_gemm_kernel<<<gemmBlocks, TB, 0, stream>>>(aggrh, selfh, w2lh, w2rh, b2l, sum); // h2 -> sum

  // ---- pool + classify ----
  zero_f32<<<cdiv(NG * DF, TB), TB, 0, stream>>>(pooled, NG * DF);
  pool_kernel<<<poolBlocks, TB, 0, stream>>>(sum, bat, pooled);
  cls_kernel<<<cdiv(NG * DOUT, TB), TB, 0, stream>>>(pooled, wcls, bcls, out);
}
